// HeGGANoAttn_44392781971734
// MI455X (gfx1250) — matrix operand order repaired and checked by reading, not compiled, not run on hardware
//
#include <hip/hip_runtime.h>

#define N_NODES 20000
#define N_EDGES 320000
#define HID 128
#define T_TYPES 3
#define LAYERS 3
#define RT 8                 // row-tiles (of 16 rows) per block: 128 rows/block

typedef __attribute__((ext_vector_type(16))) __bf16 v16bf;
typedef __attribute__((ext_vector_type(8)))  float  v8f;

__device__ __forceinline__ unsigned short f2bf(float f) {
  unsigned int u = __builtin_bit_cast(unsigned int, f);
  unsigned int r = u + 0x7FFFu + ((u >> 16) & 1u);   // round-to-nearest-even
  return (unsigned short)(r >> 16);
}
__device__ __forceinline__ float bf2f(unsigned short h) {
  unsigned int u = ((unsigned int)h) << 16;
  return __builtin_bit_cast(float, u);
}

// ---------------------------------------------------------------------------
// Weight prep: f32 [K x ldw] -> bf16 WMMA B-fragments.
// Layout [chunk][ntile][lane][e]; ISA B-layout (16-bit, 32x16):
//   half h = lane>>4: e<8 holds k = 8h + e, e>=8 holds k = 16 + 8h + (e-8), col = lane & 15.
// ---------------------------------------------------------------------------
__global__ void k_prep(const float* __restrict__ W, unsigned short* __restrict__ frag,
                       int K, int ldw, int ncols, int chunks, int ntiles) {
  int total = chunks * ntiles * 512;
  for (int i = blockIdx.x * 256 + threadIdx.x; i < total; i += gridDim.x * 256) {
    int e    = i & 15;
    int lane = (i >> 4) & 31;
    int rest = i >> 9;
    int nt   = rest % ntiles;
    int c    = rest / ntiles;
    int kk   = ((e >> 3) << 4) | ((lane >> 4) << 3) | (e & 7);
    int k    = (c << 5) + kk;
    int col  = (nt << 4) + (lane & 15);
    float v  = (k < K && col < ncols) ? W[k * ldw + col] : 0.f;
    frag[i]  = f2bf(v);
  }
}

// ---------------------------------------------------------------------------
// A-tile element fetch by assembly mode.
// mode 0: bf16 buffer rows x K
// mode 1: concat(x f32 [N,6], pe f32 [N,8])  (K=14, zero-padded)
// mode 2: edge_attr f32 [E,4]                (K=4, zero-padded)
// mode 3: concat(hn[src[e]], hn[dst[e]], he[e])  all bf16 x128  (K=384)
// mode 4: concat(hn[row] bf16 x128, m_node f32 x128)            (K=256)
// ---------------------------------------------------------------------------
__device__ __forceinline__ float fetchA(int mode, const void* s0v, const void* s1v,
                                        const int* idx0, const int* idx1,
                                        int row, int k, int K) {
  if (k >= K) return 0.f;
  switch (mode) {
    case 0: return bf2f(((const unsigned short*)s0v)[row * K + k]);
    case 1: {
      if (k < 6) return ((const float*)s0v)[row * 6 + k];
      return ((const float*)s1v)[row * 8 + (k - 6)];
    }
    case 2: return ((const float*)s0v)[row * 4 + k];
    case 3: {
      const unsigned short* hn = (const unsigned short*)s0v;
      const unsigned short* he = (const unsigned short*)s1v;
      if (k < 128) return bf2f(hn[idx0[row] * 128 + k]);
      if (k < 256) return bf2f(hn[idx1[row] * 128 + (k - 128)]);
      return bf2f(he[row * 128 + (k - 256)]);
    }
    default: {
      if (k < 128) return bf2f(((const unsigned short*)s0v)[row * 128 + k]);
      return ((const float*)s1v)[row * 128 + (k - 128)];
    }
  }
}

// ---------------------------------------------------------------------------
// Fused row-MLP: out[128 rows x 128 cols] = act(A @ W + b) (+res), type-masked.
// 256 threads = 8 wave32; wave w owns cols [16w,16w+16) for all 8 row tiles.
// Per K-chunk: one B-fragment load feeds 8 WMMAs with independent accumulators.
// Software pipelining: A(t+1) prefetched past WMMA(t); B(c+1) prefetched past
// the 8-WMMA sweep of chunk c, so ds/vmem latency hides behind the XDL chain.
// ---------------------------------------------------------------------------
__global__ void __launch_bounds__(256) k_mlp(
    int mode, const void* s0v, const void* s1v, const int* idx0, const int* idx1,
    int rows, int K,
    const unsigned short* __restrict__ Wfrag, const float* __restrict__ bias,
    int relu, const unsigned short* resid,
    const int* types, int tsel,
    unsigned short* out_bf16, float* out_f32, int ldc, int ncols, int ntiles) {
  extern __shared__ __align__(32) unsigned short Afrag[];  // RT*chunks*512 shorts
  const int chunks = (K + 31) >> 5;
  const int row0 = blockIdx.x * (RT * 16);

  // ---- stage A tiles (RT*16 rows x chunks*32) into LDS in fragment order
  const int pairsPerRow = chunks << 4;                 // (chunks*32)/2
  const int totalPairs  = pairsPerRow * (RT * 16);
  for (int p = threadIdx.x; p < totalPairs; p += 256) {
    int r  = p / pairsPerRow;                          // 0 .. RT*16-1
    int k0 = (p - r * pairsPerRow) << 1;               // even k
    int row = row0 + r;
    float v0 = 0.f, v1 = 0.f;
    if (row < rows) {
      v0 = fetchA(mode, s0v, s1v, idx0, idx1, row, k0, K);
      v1 = fetchA(mode, s0v, s1v, idx0, idx1, row, k0 + 1, K);
    }
    int t = r >> 4, rl = r & 15;
    int c = k0 >> 5, kk = k0 & 31;
    int laneHi = (kk >> 3) & 1;                        // kk in [8,16)|[24,32)
    int lane = rl | (laneHi << 4);
    int e = laneHi ? ((kk < 16) ? kk - 8 : kk - 16)
                   : ((kk < 8) ? kk : kk - 8);
    unsigned int pack = (unsigned int)f2bf(v0) | ((unsigned int)f2bf(v1) << 16);
    *(unsigned int*)&Afrag[((((t * chunks + c) << 5) | lane) << 4) + e] = pack;
  }
  __syncthreads();

  // ---- WMMA main loop with A/B double-buffering
  const int wave = threadIdx.x >> 5;
  const int lane = threadIdx.x & 31;
  if (wave < ntiles) {
    auto loadA = [&](int t, int c) -> v16bf {
      return *(const v16bf*)&Afrag[(((t * chunks + c) << 5) | lane) << 4];
    };
    auto loadB = [&](int c) -> v16bf {
      return *(const v16bf*)&Wfrag[(size_t)(((c * ntiles + wave) << 5) | lane) << 4];
    };
    v8f zero = {};
    v8f acc[RT];
#pragma unroll
    for (int t = 0; t < RT; ++t) acc[t] = zero;

    v16bf b = loadB(0);
    for (int c = 0; c < chunks; ++c) {
      v16bf bn = (c + 1 < chunks) ? loadB(c + 1) : b;  // prefetch next B (global)
      v16bf a = loadA(0, c);
#pragma unroll
      for (int t = 0; t < RT; ++t) {
        v16bf an = (t + 1 < RT) ? loadA(t + 1, c) : a; // prefetch next A (LDS)
        acc[t] = __builtin_amdgcn_wmma_f32_16x16x32_bf16(false, a, false, b, (short)0,
                                                         acc[t], false, false);
        a = an;
      }
      b = bn;
    }

    // ---- epilogue: bias, relu, residual, typed write-mask
    int col   = (wave << 4) | (lane & 15);
    int rBase = (lane >> 4) << 3;                      // lanes>=16 hold rows 8..15
    if (col < ncols) {
      float bcol = bias[col];
#pragma unroll
      for (int t = 0; t < RT; ++t) {
#pragma unroll
        for (int v = 0; v < 8; ++v) {
          int row = row0 + (t << 4) + rBase + v;
          if (row >= rows) continue;
          if (types && types[row] != tsel) continue;
          float val = acc[t][v] + bcol;
          if (relu) val = fmaxf(val, 0.f);
          if (resid) val += bf2f(resid[row * 128 + col]);
          if (out_bf16) out_bf16[row * 128 + col] = f2bf(val);
          else          out_f32[row * ldc + col]  = val;
        }
      }
    }
  }
}

__global__ void k_zero(float* p, int n) {
  int i = blockIdx.x * 256 + threadIdx.x;
  if (i < n) p[i] = 0.f;
}

// scatter-add h_edges (bf16) into m_node (f32) via dst index; HW f32 atomics.
__global__ void k_scatter(const unsigned short* __restrict__ he, const int* __restrict__ dst,
                          float* mnode, int E) {
  int idx = blockIdx.x * 256 + threadIdx.x;
  int e = idx >> 5;
  if (e >= E) return;
  int c0 = (idx & 31) << 2;
  int d = dst[e];
  const unsigned short* p = &he[e * 128 + c0];
  float* q = &mnode[(size_t)d * 128 + c0];
#pragma unroll
  for (int j = 0; j < 4; ++j)
    __hip_atomic_fetch_add(&q[j], bf2f(p[j]), __ATOMIC_RELAXED, __HIP_MEMORY_SCOPE_AGENT);
}

// ---------------------------------------------------------------------------
extern "C" void kernel_launch(void* const* d_in, const int* in_sizes, int n_in,
                              void* d_out, int out_size, void* d_ws, size_t ws_size,
                              hipStream_t stream) {
  const float* x     = (const float*)d_in[0];
  const float* pe    = (const float*)d_in[1];
  const float* ea    = (const float*)d_in[2];
  const float* encW1 = (const float*)d_in[3];
  const float* encB1 = (const float*)d_in[4];
  const float* encW2 = (const float*)d_in[5];
  const float* encB2 = (const float*)d_in[6];
  const float* eeW1  = (const float*)d_in[7];
  const float* eeB1  = (const float*)d_in[8];
  const float* eeW2  = (const float*)d_in[9];
  const float* eeB2  = (const float*)d_in[10];
  const float* euW1  = (const float*)d_in[11];
  const float* euB1  = (const float*)d_in[12];
  const float* euW2  = (const float*)d_in[13];
  const float* euB2  = (const float*)d_in[14];
  const float* nuW1  = (const float*)d_in[15];
  const float* nuB1  = (const float*)d_in[16];
  const float* nuW2  = (const float*)d_in[17];
  const float* nuB2  = (const float*)d_in[18];
  const float* fuW1  = (const float*)d_in[19];
  const float* fuB1  = (const float*)d_in[20];
  const float* fuW2  = (const float*)d_in[21];
  const float* fuB2  = (const float*)d_in[22];
  const float* decW1 = (const float*)d_in[23];
  const float* decB1 = (const float*)d_in[24];
  const float* decW2 = (const float*)d_in[25];
  const float* decB2 = (const float*)d_in[26];
  const int*   eidx  = (const int*)d_in[27];
  const int*   ntype = (const int*)d_in[28];
  const int* src = eidx;
  const int* dst = eidx + N_EDGES;
  float* out = (float*)d_out;

  size_t off = 0;
  auto alloc = [&](size_t bytes) -> char* {
    char* p = (char*)d_ws + off;
    off += (bytes + 255) & ~(size_t)255;
    return p;
  };

  // fragment element counts: chunks*ntiles*512 (bf16 shorts)
  const int FE1 = 1 * 8 * 512, FE2 = 4 * 8 * 512;        // K<=32 / 128x128
  const int FEU1 = 12 * 8 * 512, FNU1 = 8 * 8 * 512;     // K=384 / K=256
  const int FD2 = 4 * 1 * 512;                           // dec l2 (128->4)

  unsigned short* fEnc1 = (unsigned short*)alloc((size_t)3 * FE1 * 2);
  unsigned short* fEnc2 = (unsigned short*)alloc((size_t)3 * FE2 * 2);
  unsigned short* fEe1  = (unsigned short*)alloc((size_t)FE1 * 2);
  unsigned short* fEe2  = (unsigned short*)alloc((size_t)FE2 * 2);
  unsigned short* fEu1  = (unsigned short*)alloc((size_t)3 * FEU1 * 2);
  unsigned short* fEu2  = (unsigned short*)alloc((size_t)3 * FE2 * 2);
  unsigned short* fNu1  = (unsigned short*)alloc((size_t)3 * FNU1 * 2);
  unsigned short* fNu2  = (unsigned short*)alloc((size_t)3 * FE2 * 2);
  unsigned short* fFu1  = (unsigned short*)alloc((size_t)3 * FE2 * 2);
  unsigned short* fFu2  = (unsigned short*)alloc((size_t)3 * FE2 * 2);
  unsigned short* fDec1 = (unsigned short*)alloc((size_t)3 * FE2 * 2);
  unsigned short* fDec2 = (unsigned short*)alloc((size_t)3 * FD2 * 2);

  unsigned short* hNodes = (unsigned short*)alloc((size_t)N_NODES * 128 * 2);
  unsigned short* tmpN   = (unsigned short*)alloc((size_t)N_NODES * 128 * 2);
  unsigned short* localN = (unsigned short*)alloc((size_t)N_NODES * 128 * 2);
  unsigned short* hEdges = (unsigned short*)alloc((size_t)N_EDGES * 128 * 2);
  unsigned short* tmpE   = (unsigned short*)alloc((size_t)N_EDGES * 128 * 2);
  float*          mNode  = (float*)alloc((size_t)N_NODES * 128 * 4);
  (void)ws_size; (void)in_sizes; (void)n_in; (void)out_size;

  auto prep = [&](const float* W, unsigned short* frag, int K, int ldw, int ncols,
                  int chunks, int ntiles) {
    int total = chunks * ntiles * 512;
    k_prep<<<(total + 255) / 256, 256, 0, stream>>>(W, frag, K, ldw, ncols, chunks, ntiles);
  };
  auto mlp = [&](int mode, const void* s0, const void* s1, const int* i0, const int* i1,
                 int rows, int K, const unsigned short* Wf, const float* b, int relu,
                 const unsigned short* resid, const int* types, int tsel,
                 unsigned short* ob, float* of, int ldc, int ncols, int ntiles) {
    int chunks = (K + 31) >> 5;
    size_t ldsBytes = (size_t)RT * chunks * 512 * 2;   // up to 96 KB (K=384)
    int grid = (rows + RT * 16 - 1) / (RT * 16);
    k_mlp<<<grid, 256, ldsBytes, stream>>>(mode, s0, s1, i0, i1, rows, K, Wf, b,
                                           relu, resid, types, tsel, ob, of, ldc,
                                           ncols, ntiles);
  };

  // ---- weight prep (bf16 fragment swizzle) ----
  for (int t = 0; t < T_TYPES; ++t) {
    prep(encW1 + (size_t)t * 14 * 128, fEnc1 + (size_t)t * FE1, 14, 128, 128, 1, 8);
    prep(encW2 + (size_t)t * 128 * 128, fEnc2 + (size_t)t * FE2, 128, 128, 128, 4, 8);
    prep(decW1 + (size_t)t * 128 * 128, fDec1 + (size_t)t * FE2, 128, 128, 128, 4, 8);
    prep(decW2 + (size_t)t * 128 * 4, fDec2 + (size_t)t * FD2, 128, 4, 4, 4, 1);
  }
  prep(eeW1, fEe1, 4, 128, 128, 1, 8);
  prep(eeW2, fEe2, 128, 128, 128, 4, 8);
  for (int l = 0; l < LAYERS; ++l) {
    prep(euW1 + (size_t)l * 384 * 128, fEu1 + (size_t)l * FEU1, 384, 128, 128, 12, 8);
    prep(euW2 + (size_t)l * 128 * 128, fEu2 + (size_t)l * FE2, 128, 128, 128, 4, 8);
    prep(nuW1 + (size_t)l * 256 * 128, fNu1 + (size_t)l * FNU1, 256, 128, 128, 8, 8);
    prep(nuW2 + (size_t)l * 128 * 128, fNu2 + (size_t)l * FE2, 128, 128, 128, 4, 8);
    prep(fuW1 + (size_t)l * 128 * 128, fFu1 + (size_t)l * FE2, 128, 128, 128, 4, 8);
    prep(fuW2 + (size_t)l * 128 * 128, fFu2 + (size_t)l * FE2, 128, 128, 128, 4, 8);
  }

  // ---- typed node encode: h_nodes = select_t( mlp_t([x||pe]) ) ----
  for (int t = 0; t < T_TYPES; ++t) {
    mlp(1, x, pe, nullptr, nullptr, N_NODES, 14, fEnc1 + (size_t)t * FE1, encB1 + t * 128,
        1, nullptr, nullptr, 0, tmpN, nullptr, 128, 128, 8);
    mlp(0, tmpN, nullptr, nullptr, nullptr, N_NODES, 128, fEnc2 + (size_t)t * FE2,
        encB2 + t * 128, 0, nullptr, ntype, t, hNodes, nullptr, 128, 128, 8);
  }
  // ---- edge encode ----
  mlp(2, ea, nullptr, nullptr, nullptr, N_EDGES, 4, fEe1, eeB1, 1, nullptr, nullptr, 0,
      tmpE, nullptr, 128, 128, 8);
  mlp(0, tmpE, nullptr, nullptr, nullptr, N_EDGES, 128, fEe2, eeB2, 0, nullptr, nullptr, 0,
      hEdges, nullptr, 128, 128, 8);

  // ---- message-passing layers ----
  for (int l = 0; l < LAYERS; ++l) {
    // edge update: h_edges += MLP([hn[src] || hn[dst] || h_edges])
    mlp(3, hNodes, hEdges, src, dst, N_EDGES, 384, fEu1 + (size_t)l * FEU1, euB1 + l * 128,
        1, nullptr, nullptr, 0, tmpE, nullptr, 128, 128, 8);
    mlp(0, tmpE, nullptr, nullptr, nullptr, N_EDGES, 128, fEu2 + (size_t)l * FE2,
        euB2 + l * 128, 0, hEdges, nullptr, 0, hEdges, nullptr, 128, 128, 8);
    // aggregate to destination nodes
    k_zero<<<(N_NODES * 128 + 255) / 256, 256, 0, stream>>>(mNode, N_NODES * 128);
    k_scatter<<<(N_EDGES * 32 + 255) / 256, 256, 0, stream>>>(hEdges, dst, mNode, N_EDGES);
    // node update: local = h_nodes + MLP([h_nodes || m_node]); h_nodes = MLP(local)
    mlp(4, hNodes, mNode, nullptr, nullptr, N_NODES, 256, fNu1 + (size_t)l * FNU1,
        nuB1 + l * 128, 1, nullptr, nullptr, 0, tmpN, nullptr, 128, 128, 8);
    mlp(0, tmpN, nullptr, nullptr, nullptr, N_NODES, 128, fNu2 + (size_t)l * FE2,
        nuB2 + l * 128, 0, hNodes, nullptr, 0, localN, nullptr, 128, 128, 8);
    mlp(0, localN, nullptr, nullptr, nullptr, N_NODES, 128, fFu1 + (size_t)l * FE2,
        fuB1 + l * 128, 1, nullptr, nullptr, 0, tmpN, nullptr, 128, 128, 8);
    mlp(0, tmpN, nullptr, nullptr, nullptr, N_NODES, 128, fFu2 + (size_t)l * FE2,
        fuB2 + l * 128, 0, nullptr, nullptr, 0, hNodes, nullptr, 128, 128, 8);
  }

  // ---- typed decode -> d_out [N,4] f32 ----
  for (int t = 0; t < T_TYPES; ++t) {
    mlp(0, hNodes, nullptr, nullptr, nullptr, N_NODES, 128, fDec1 + (size_t)t * FE2,
        decB1 + t * 128, 1, nullptr, nullptr, 0, tmpN, nullptr, 128, 128, 8);
    mlp(0, tmpN, nullptr, nullptr, nullptr, N_NODES, 128, fDec2 + (size_t)t * FD2,
        decB2 + t * 4, 0, nullptr, ntype, t, nullptr, out, 4, 4, 1);
  }
}
